// DGRUCell_59210419143195
// MI455X (gfx1250) — matrix operand order repaired and checked by reading, not compiled, not run on hardware
//
#include <hip/hip_runtime.h>
#include <cstdint>

typedef unsigned short ushort_t;
typedef __attribute__((ext_vector_type(16))) __bf16 v16bf;
typedef __attribute__((ext_vector_type(8)))  float  v8f;
typedef __attribute__((ext_vector_type(4)))  int    v4i;

#define AS1 __attribute__((address_space(1)))
#define AS3 __attribute__((address_space(3)))

#define B_ROWS 8192
#define D_DIM  1024
#define KDIM   2048
#define MT     64
#define NT     64
#define KT     32
#define PAD    40              // LDS row stride in bf16 elems (80B, conflict-free for b128 frags)
#define NKSTEP (KDIM / KT)     // 64

#if __has_builtin(__builtin_amdgcn_global_load_async_to_lds_b128) && \
    __has_builtin(__builtin_amdgcn_s_wait_asynccnt)
#define USE_ASYNC 1
#else
#define USE_ASYNC 0
#endif

union Frag { v16bf bf; uint4 q[2]; };

__device__ __forceinline__ ushort_t f2bf(float f) {
    uint32_t u = __float_as_uint(f);
    uint32_t r = u + 0x7FFFu + ((u >> 16) & 1u);   // round-to-nearest-even
    return (ushort_t)(r >> 16);
}
__device__ __forceinline__ float bf2f(ushort_t s) {
    return __uint_as_float(((uint32_t)s) << 16);
}

// ---------------------------------------------------------------- fp32 -> bf16 weight convert
__global__ __launch_bounds__(256) void cvt_weights_kernel(
    const float* __restrict__ a, ushort_t* __restrict__ oa, int na,
    const float* __restrict__ b, ushort_t* __restrict__ ob, int nb)
{
    int stride = gridDim.x * blockDim.x;
    for (int i = blockIdx.x * blockDim.x + threadIdx.x; i < na + nb; i += stride) {
        if (i < na) oa[i] = f2bf(a[i]);
        else        ob[i - na] = f2bf(b[i - na]);
    }
}

// ---------------------------------------------------------------- LN1: LayerNorm(concat(x,h)) -> bf16
__global__ __launch_bounds__(256) void ln1_kernel(
    const float* __restrict__ x, const float* __restrict__ h,
    const float* __restrict__ w, const float* __restrict__ b,
    ushort_t* __restrict__ out)
{
    const int row = blockIdx.x;
    const int t = threadIdx.x;
    float v[8];
    float s = 0.f, s2 = 0.f;
#pragma unroll
    for (int i = 0; i < 8; ++i) {
        int j = t + i * 256;
        float val = (j < D_DIM) ? x[(size_t)row * D_DIM + j]
                                : h[(size_t)row * D_DIM + (j - D_DIM)];
        v[i] = val; s += val; s2 += val * val;
    }
#pragma unroll
    for (int off = 16; off > 0; off >>= 1) {
        s  += __shfl_xor(s,  off, 32);
        s2 += __shfl_xor(s2, off, 32);
    }
    __shared__ float red[16];
    const int wave = t >> 5, lane = t & 31;
    if (lane == 0) { red[wave] = s; red[8 + wave] = s2; }
    __syncthreads();
    float ts = 0.f, ts2 = 0.f;
#pragma unroll
    for (int i = 0; i < 8; ++i) { ts += red[i]; ts2 += red[8 + i]; }
    const float mu = ts * (1.f / 2048.f);
    const float var = ts2 * (1.f / 2048.f) - mu * mu;
    const float rstd = rsqrtf(var + 1e-5f);
#pragma unroll
    for (int i = 0; i < 8; ++i) {
        int j = t + i * 256;
        out[(size_t)row * KDIM + j] = f2bf((v[i] - mu) * rstd * w[j] + b[j]);
    }
}

// ---------------------------------------------------------------- LN2: in-place LayerNorm on bf16 rows
__global__ __launch_bounds__(256) void ln2_kernel(
    ushort_t* __restrict__ io, const float* __restrict__ w, const float* __restrict__ b)
{
    const int row = blockIdx.x;
    const int t = threadIdx.x;
    float v[8];
    float s = 0.f, s2 = 0.f;
#pragma unroll
    for (int i = 0; i < 8; ++i) {
        int j = t + i * 256;
        float val = bf2f(io[(size_t)row * KDIM + j]);
        v[i] = val; s += val; s2 += val * val;
    }
#pragma unroll
    for (int off = 16; off > 0; off >>= 1) {
        s  += __shfl_xor(s,  off, 32);
        s2 += __shfl_xor(s2, off, 32);
    }
    __shared__ float red[16];
    const int wave = t >> 5, lane = t & 31;
    if (lane == 0) { red[wave] = s; red[8 + wave] = s2; }
    __syncthreads();
    float ts = 0.f, ts2 = 0.f;
#pragma unroll
    for (int i = 0; i < 8; ++i) { ts += red[i]; ts2 += red[8 + i]; }
    const float mu = ts * (1.f / 2048.f);
    const float var = ts2 * (1.f / 2048.f) - mu * mu;
    const float rstd = rsqrtf(var + 1e-5f);
#pragma unroll
    for (int i = 0; i < 8; ++i) {
        int j = t + i * 256;
        io[(size_t)row * KDIM + j] = f2bf((v[i] - mu) * rstd * w[j] + b[j]);
    }
}

// ---------------------------------------------------------------- GEMM1: gates = ln1 @ Wg^T, fused gate epilogue
__global__ __launch_bounds__(256) void gemm_gates_kernel(
    const ushort_t* __restrict__ A,    // ln1 bf16 [8192][2048]
    const ushort_t* __restrict__ Wg,   // bf16 [5120][2048]
    const float* __restrict__ bg,      // [5120]
    const float* __restrict__ x,
    const float* __restrict__ h,
    ushort_t* __restrict__ inp2,       // bf16 [8192][2048]
    float* __restrict__ z2buf,         // [8192][1024]
    float* __restrict__ out)           // [8192][1024]
{
    __shared__ ushort_t lds[2][(64 + 5 * 64) * PAD];   // 61440 bytes
    const int t = threadIdx.x;
    const int m0 = blockIdx.x * MT;
    const int n0 = blockIdx.y * NT;
    const int wave = t >> 5, lane = t & 31;
    const int wm = wave & 3;                           // 0..3 : 16-row M slice
    const int wN = wave >> 2;                          // 0..1 : which 32-wide N half
    const int hh = lane >> 4, lm = lane & 15;

    // staging addresses (per-thread, fixed; add k0 each step)
    const int arow = t >> 2, ac = t & 3;
    const ushort_t* gA = A + (size_t)(m0 + arow) * KDIM + ac * 8;
    const int aoff = arow * PAD + ac * 8;
    const ushort_t* gW[5];
    int woff[5];
#pragma unroll
    for (int j = 0; j < 5; ++j) {
        int idx = t + 256 * j;                 // 0..1279
        int wrow = idx >> 2, c = idx & 3;      // 0..319
        int gate = wrow >> 6, r = wrow & 63;
        gW[j] = Wg + (size_t)(gate * D_DIM + n0 + r) * KDIM + c * 8;
        woff[j] = (64 + wrow) * PAD + c * 8;
    }

    const v8f vzero = {0.f, 0.f, 0.f, 0.f, 0.f, 0.f, 0.f, 0.f};
    v8f acc[5][2];
#pragma unroll
    for (int g = 0; g < 5; ++g)
        for (int n = 0; n < 2; ++n) acc[g][n] = vzero;

    auto compute = [&](int cur) {
        const ushort_t* Lb = &lds[cur][0];
        Frag fa;
        const int abase = (wm * 16 + lm) * PAD;
        fa.q[0] = *(const uint4*)&Lb[abase + 8 * hh];
        fa.q[1] = *(const uint4*)&Lb[abase + 16 + 8 * hh];
#pragma unroll
        for (int g = 0; g < 5; ++g) {
#pragma unroll
            for (int n = 0; n < 2; ++n) {
                const int brow = 64 + g * 64 + wN * 32 + n * 16 + lm;
                Frag fb;
                fb.q[0] = *(const uint4*)&Lb[brow * PAD + 8 * hh];
                fb.q[1] = *(const uint4*)&Lb[brow * PAD + 16 + 8 * hh];
                acc[g][n] = __builtin_amdgcn_wmma_f32_16x16x32_bf16(
                    false, fa.bf, false, fb.bf, (short)0, acc[g][n], false, false);
            }
        }
    };

#if USE_ASYNC
    auto stage = [&](int ks, int nb) {
        const int k0 = ks * KT;
        __builtin_amdgcn_global_load_async_to_lds_b128(
            (AS1 v4i*)(gA + k0), (AS3 v4i*)&lds[nb][aoff], 0, 0);
#pragma unroll
        for (int j = 0; j < 5; ++j)
            __builtin_amdgcn_global_load_async_to_lds_b128(
                (AS1 v4i*)(gW[j] + k0), (AS3 v4i*)&lds[nb][woff[j]], 0, 0);
    };
    stage(0, 0);
    __builtin_amdgcn_s_wait_asynccnt(0);
    __syncthreads();
    for (int ks = 0; ks < NKSTEP; ++ks) {
        const int cur = ks & 1;
        if (ks + 1 < NKSTEP) stage(ks + 1, cur ^ 1);
        compute(cur);
        __builtin_amdgcn_s_wait_asynccnt(0);
        __syncthreads();
    }
#else
    uint4 ra, rw[5];
    auto load_regs = [&](int ks) {
        const int k0 = ks * KT;
        ra = *(const uint4*)(gA + k0);
#pragma unroll
        for (int j = 0; j < 5; ++j) rw[j] = *(const uint4*)(gW[j] + k0);
    };
    auto store_regs = [&](int nb) {
        *(uint4*)&lds[nb][aoff] = ra;
#pragma unroll
        for (int j = 0; j < 5; ++j) *(uint4*)&lds[nb][woff[j]] = rw[j];
    };
    load_regs(0);
    store_regs(0);
    __syncthreads();
    for (int ks = 0; ks < NKSTEP; ++ks) {
        const int cur = ks & 1;
        if (ks + 1 < NKSTEP) load_regs(ks + 1);
        compute(cur);
        if (ks + 1 < NKSTEP) store_regs(cur ^ 1);
        __syncthreads();
    }
#endif

    // epilogue: bias + sigmoid + softmax, write inp2 / z2 / partial out
#pragma unroll
    for (int n = 0; n < 2; ++n) {
#pragma unroll
        for (int r = 0; r < 8; ++r) {
            const int mg = m0 + wm * 16 + 8 * hh + r;
            const int ng = n0 + wN * 32 + n * 16 + lm;
            float g0 = acc[0][n][r] + bg[ng];
            float g1 = acc[1][n][r] + bg[1024 + ng];
            float g2 = acc[2][n][r] + bg[2048 + ng];
            float g3 = acc[3][n][r] + bg[3072 + ng];
            float g4 = acc[4][n][r] + bg[4096 + ng];
            float rx = 1.f / (1.f + __expf(-g0));
            float rh = 1.f / (1.f + __expf(-g1));
            float mx = fmaxf(g2, fmaxf(g3, g4));
            float e0 = __expf(g2 - mx), e1 = __expf(g3 - mx), e2 = __expf(g4 - mx);
            float rs = 1.f / (e0 + e1 + e2);
            float z0 = e0 * rs, z1 = e1 * rs, z2v = e2 * rs;
            const size_t o = (size_t)mg * D_DIM + ng;
            float xv = x[o], hv = h[o];
            inp2[(size_t)mg * KDIM + ng] = f2bf(xv * rx);
            inp2[(size_t)mg * KDIM + D_DIM + ng] = f2bf(hv * rh);
            out[o] = z0 * xv + z1 * hv;
            z2buf[o] = z2v;
        }
    }
}

// ---------------------------------------------------------------- GEMM2: u = tanh(ln2 @ Wu^T + bu); out += z2*u
__global__ __launch_bounds__(256) void gemm_u_kernel(
    const ushort_t* __restrict__ A,    // ln2(inp2) bf16 [8192][2048]
    const ushort_t* __restrict__ Wu,   // bf16 [1024][2048]
    const float* __restrict__ bu,      // [1024]
    const float* __restrict__ z2buf,   // [8192][1024]
    float* __restrict__ out)           // [8192][1024]
{
    __shared__ ushort_t lds[2][(64 + 64) * PAD];       // 20480 bytes
    const int t = threadIdx.x;
    const int m0 = blockIdx.x * MT;
    const int n0 = blockIdx.y * NT;
    const int wave = t >> 5, lane = t & 31;
    const int wm = wave & 3, wN = wave >> 2;
    const int hh = lane >> 4, lm = lane & 15;

    const int arow = t >> 2, ac = t & 3;
    const ushort_t* gA = A + (size_t)(m0 + arow) * KDIM + ac * 8;
    const int aoff = arow * PAD + ac * 8;
    const ushort_t* gW = Wu + (size_t)(n0 + arow) * KDIM + ac * 8;
    const int woff = (64 + arow) * PAD + ac * 8;

    const v8f vzero = {0.f, 0.f, 0.f, 0.f, 0.f, 0.f, 0.f, 0.f};
    v8f acc[2] = {vzero, vzero};

    auto compute = [&](int cur) {
        const ushort_t* Lb = &lds[cur][0];
        Frag fa;
        const int abase = (wm * 16 + lm) * PAD;
        fa.q[0] = *(const uint4*)&Lb[abase + 8 * hh];
        fa.q[1] = *(const uint4*)&Lb[abase + 16 + 8 * hh];
#pragma unroll
        for (int n = 0; n < 2; ++n) {
            const int brow = 64 + wN * 32 + n * 16 + lm;
            Frag fb;
            fb.q[0] = *(const uint4*)&Lb[brow * PAD + 8 * hh];
            fb.q[1] = *(const uint4*)&Lb[brow * PAD + 16 + 8 * hh];
            acc[n] = __builtin_amdgcn_wmma_f32_16x16x32_bf16(
                false, fa.bf, false, fb.bf, (short)0, acc[n], false, false);
        }
    };

#if USE_ASYNC
    auto stage = [&](int ks, int nb) {
        const int k0 = ks * KT;
        __builtin_amdgcn_global_load_async_to_lds_b128(
            (AS1 v4i*)(gA + k0), (AS3 v4i*)&lds[nb][aoff], 0, 0);
        __builtin_amdgcn_global_load_async_to_lds_b128(
            (AS1 v4i*)(gW + k0), (AS3 v4i*)&lds[nb][woff], 0, 0);
    };
    stage(0, 0);
    __builtin_amdgcn_s_wait_asynccnt(0);
    __syncthreads();
    for (int ks = 0; ks < NKSTEP; ++ks) {
        const int cur = ks & 1;
        if (ks + 1 < NKSTEP) stage(ks + 1, cur ^ 1);
        compute(cur);
        __builtin_amdgcn_s_wait_asynccnt(0);
        __syncthreads();
    }
#else
    uint4 ra, rw;
    auto load_regs = [&](int ks) {
        const int k0 = ks * KT;
        ra = *(const uint4*)(gA + k0);
        rw = *(const uint4*)(gW + k0);
    };
    auto store_regs = [&](int nb) {
        *(uint4*)&lds[nb][aoff] = ra;
        *(uint4*)&lds[nb][woff] = rw;
    };
    load_regs(0);
    store_regs(0);
    __syncthreads();
    for (int ks = 0; ks < NKSTEP; ++ks) {
        const int cur = ks & 1;
        if (ks + 1 < NKSTEP) load_regs(ks + 1);
        compute(cur);
        if (ks + 1 < NKSTEP) store_regs(cur ^ 1);
        __syncthreads();
    }
#endif

#pragma unroll
    for (int n = 0; n < 2; ++n) {
#pragma unroll
        for (int r = 0; r < 8; ++r) {
            const int mg = m0 + wm * 16 + 8 * hh + r;
            const int ng = n0 + wN * 32 + n * 16 + lm;
            const size_t o = (size_t)mg * D_DIM + ng;
            float uv = tanhf(acc[n][r] + bu[ng]);
            out[o] = out[o] + z2buf[o] * uv;
        }
    }
}

// ---------------------------------------------------------------- launch
extern "C" void kernel_launch(void* const* d_in, const int* in_sizes, int n_in,
                              void* d_out, int out_size, void* d_ws, size_t ws_size,
                              hipStream_t stream) {
    (void)in_sizes; (void)n_in; (void)out_size; (void)ws_size;
    const float* x    = (const float*)d_in[0];
    const float* h    = (const float*)d_in[1];
    const float* ln_w = (const float*)d_in[2];
    const float* ln_b = (const float*)d_in[3];
    const float* ln2w = (const float*)d_in[4];
    const float* ln2b = (const float*)d_in[5];
    const float* Wg   = (const float*)d_in[6];
    const float* bg   = (const float*)d_in[7];
    const float* Wu   = (const float*)d_in[8];
    const float* bu   = (const float*)d_in[9];
    float* out = (float*)d_out;

    char* ws = (char*)d_ws;
    const size_t WG_N = (size_t)5 * D_DIM * KDIM;   // 10,485,760
    const size_t WU_N = (size_t)D_DIM * KDIM;       //  2,097,152
    ushort_t* wgbf = (ushort_t*)(ws);                               // 20,971,520 B
    ushort_t* wubf = (ushort_t*)(ws + 20971520);                    //  4,194,304 B
    ushort_t* ln1  = (ushort_t*)(ws + 25165824);                    // 33,554,432 B
    ushort_t* inp2 = (ushort_t*)(ws + 58720256);                    // 33,554,432 B
    float*    z2b  = (float*)   (ws + 92274688);                    // 33,554,432 B

    cvt_weights_kernel<<<4096, 256, 0, stream>>>(Wg, wgbf, (int)WG_N, Wu, wubf, (int)WU_N);
    ln1_kernel<<<B_ROWS, 256, 0, stream>>>(x, h, ln_w, ln_b, ln1);
    gemm_gates_kernel<<<dim3(B_ROWS / MT, D_DIM / NT), 256, 0, stream>>>(
        ln1, wgbf, bg, x, h, inp2, z2b, out);
    ln2_kernel<<<B_ROWS, 256, 0, stream>>>(inp2, ln2w, ln2b);
    gemm_u_kernel<<<dim3(B_ROWS / MT, D_DIM / NT), 256, 0, stream>>>(
        inp2, wubf, bu, z2b, out);
}
